// Spectral_Pool_45140106281471
// MI455X (gfx1250) — compile-verified, matrix operand order checked by reading, not run on hardware
//
#include <hip/hip_runtime.h>
#include <hip/hip_bf16.h>
#include <stdint.h>

// ---------------------------------------------------------------------------
// Spectral pooling (16,256,256,64) -> (16,217,217,64) as Y = Re(A X A^T)
//   = Ar X Ar^T - Ai X Ai^T,  A = iDFT_217 o centered-crop o DFT_256
// Two chained bf16 WMMA GEMM stages, f32 accumulation, LDS-staged coalesced
// epilogues. Batch-fused (gridDim.z=16) when workspace allows.
// ---------------------------------------------------------------------------

typedef __bf16 bf16_t;
typedef __attribute__((ext_vector_type(16))) __bf16 v16bf;
typedef __attribute__((ext_vector_type(8)))  float  v8f;

#define KOUT 217          // kept spectral size (odd -> no real-forcing branch)
#define WC   16384        // W*C = 256*64 (contiguous inner matrix dimension)
#define LDST 40           // LDS tile row stride in halfwords (32 data + 8 pad)

__device__ __forceinline__ unsigned short f2bf(float f) {
  union { float f; unsigned u; } v; v.f = f;
  unsigned r = v.u + 0x7FFFu + ((v.u >> 16) & 1u);   // round-to-nearest-even
  return (unsigned short)(r >> 16);
}

// Fragment loader: 16x32 bf16 tile from LDS (row-major, stride LDST).
// A-layout (ISA 7.12.2): lane L -> row = L%16; K-chunks (L/16)*8 and 16+(L/16)*8.
__device__ __forceinline__ v16bf load_frag(const unsigned short* lds, int row0) {
  int lane = threadIdx.x & 31;
  const unsigned short* p = lds + (row0 + (lane & 15)) * LDST + ((lane >> 4) << 3);
  union { uint4 u[2]; v16bf v; } u;
  u.u[0] = *(const uint4*)(p);        // K = ko .. ko+7
  u.u[1] = *(const uint4*)(p + 16);   // K = 16+ko .. 16+ko+7
  return u.v;
}

// angle = 2*pi * ((p mod 111104)/111104); all phases are exact multiples.
__device__ __forceinline__ float angf(long long p) {
  long long q = p % 111104; if (q < 0) q += 111104;
  return (6.2831853071795864769e0f / 111104.0f) * (float)q;
}

// Build Ar/Ai (bf16, 256x256 row-major; rows >= 217 zero).
// A[m,n] = (1/K)[ e^{i(2pi n/H + 54 th)} D109 + e^{i(2pi 218n/H + 162.5 th)} D108 ]
// th = 2pi(256m-217n)/55552, DN = sin(N th/2)/sin(th/2)  (Dirichlet sums of the
// two contiguous crop-frequency runs f=g-1 (g=0..108), f=g-218 (g=109..216)).
__global__ void build_dft_mats(unsigned short* __restrict__ Ar,
                               unsigned short* __restrict__ Ai) {
  int idx = blockIdx.x * 256 + threadIdx.x;      // 65536 entries
  int m = idx >> 8, n = idx & 255;
  float ar = 0.0f, ai = 0.0f;
  if (m < KOUT) {
    long long j = 256LL * m - 217LL * n;         // theta/2 = 2*pi*j/111104
    float D1, D2;
    if (j == 0) { D1 = 109.0f; D2 = 108.0f; }
    else {
      float sh = sinf(angf(j));
      D1 = sinf(angf(109 * j)) / sh;
      D2 = sinf(angf(108 * j)) / sh;
    }
    float a1 = angf(434LL * n + 108 * j);        // 2pi n/256   + 54*theta
    float a2 = angf(94612LL * n + 325 * j);      // 2pi 218n/256 + 162.5*theta
    ar = (D1 * cosf(a1) + D2 * cosf(a2)) * (1.0f / 217.0f);
    ai = (D1 * sinf(a1) + D2 * sinf(a2)) * (1.0f / 217.0f);
  }
  Ar[idx] = f2bf(ar);
  Ai[idx] = f2bf(ai);
}

// Stage 1: Tr = Ar * X, Tin = -(Ai * X); X is f32 256 x 16384 row-major
// (rows = h). Output stored TRANSPOSED as bf16 T'[w][m1][c]. Each block owns
// exactly one w (BN == C == 64), so its output span is contiguous -> stage
// through LDS and emit coalesced b128 stores.
__global__ __launch_bounds__(256) void spool_stage1(
    const float* __restrict__ X,
    const unsigned short* __restrict__ Ar,
    const unsigned short* __restrict__ Ai,
    unsigned short* __restrict__ Tr,
    unsigned short* __restrict__ Tin,
    size_t zStride) {
  __shared__ unsigned short sAr[128 * LDST];
  __shared__ unsigned short sAi[128 * LDST];
  __shared__ unsigned short sX [64  * LDST];
  __shared__ unsigned short sOutT[128 * 72];       // transpose staging (18 KB)

  const size_t zoff = (size_t)blockIdx.z * zStride;
  X += zoff; Tr += zoff; Tin += zoff;

  const int tid = threadIdx.x;
  const int n0  = blockIdx.x * 64;     // n = w*64+c  ->  w = blockIdx.x fixed
  const int m0  = blockIdx.y * 128;    // m1 tile (padded 256)
  const int wm  = (tid >> 5) >> 1;     // wave row 0..3
  const int wn  = (tid >> 5) & 1;      // wave col 0..1

  v8f accR[2][2] = {};
  v8f accI[2][2] = {};

  for (int k0 = 0; k0 < 256; k0 += 32) {
    #pragma unroll
    for (int i = 0; i < 2; ++i) {                  // A tiles 128x32 (x2 mats)
      int idx = tid + i * 256;
      int row = idx >> 2, col = (idx & 3) << 3;
      *(uint4*)(sAr + row * LDST + col) = *(const uint4*)(Ar + (m0 + row) * 256 + k0 + col);
      *(uint4*)(sAi + row * LDST + col) = *(const uint4*)(Ai + (m0 + row) * 256 + k0 + col);
    }
    #pragma unroll
    for (int i = 0; i < 2; ++i) {                  // X tile 32(k) x 64(n): f32->bf16, transpose
      int idx = tid + i * 256;
      int kk = idx >> 4, nn = (idx & 15) << 2;
      float4 v = *(const float4*)(X + (size_t)(k0 + kk) * WC + n0 + nn);
      sX[(nn + 0) * LDST + kk] = f2bf(v.x);
      sX[(nn + 1) * LDST + kk] = f2bf(v.y);
      sX[(nn + 2) * LDST + kk] = f2bf(v.z);
      sX[(nn + 3) * LDST + kk] = f2bf(v.w);
    }
    __syncthreads();

    v16bf arf[2], aif[2], xbf[2];
    arf[0] = load_frag(sAr, wm * 32);      arf[1] = load_frag(sAr, wm * 32 + 16);
    aif[0] = load_frag(sAi, wm * 32);      aif[1] = load_frag(sAi, wm * 32 + 16);
    xbf[0] = load_frag(sX,  wn * 32);      xbf[1] = load_frag(sX,  wn * 32 + 16);

    #pragma unroll
    for (int ms = 0; ms < 2; ++ms)
      #pragma unroll
      for (int ns = 0; ns < 2; ++ns) {
        accR[ms][ns] = __builtin_amdgcn_wmma_f32_16x16x32_bf16(
            false, arf[ms], false, xbf[ns], (short)0, accR[ms][ns], false, false);
        accI[ms][ns] = __builtin_amdgcn_wmma_f32_16x16x32_bf16(
            false, aif[ms], false, xbf[ns], (short)0, accI[ms][ns], false, false);
      }
    __syncthreads();
  }

  // Epilogue: acc -> LDS (transposed) -> coalesced b128 stores of the block's
  // contiguous T'[w][m0..m0+127][0..63] span.  (C/D frag: m = r + 8*(lane/16),
  // n = lane%16.)
  const int lane = tid & 31;
  const int rAdd = (lane >> 4) << 3;
  const int w    = blockIdx.x;
  for (int mat = 0; mat < 2; ++mat) {
    #pragma unroll
    for (int ms = 0; ms < 2; ++ms)
      #pragma unroll
      for (int ns = 0; ns < 2; ++ns) {
        int mloc = wm * 32 + ms * 16 + rAdd;
        int c    = wn * 32 + ns * 16 + (lane & 15);
        #pragma unroll
        for (int r = 0; r < 8; ++r) {
          float v = mat ? -accI[ms][ns][r] : accR[ms][ns][r];
          sOutT[(mloc + r) * 72 + c] = f2bf(v);    // pre-negated imag for mat=1
        }
      }
    __syncthreads();
    unsigned short* Tdst = (mat ? Tin : Tr) + (w << 14) + m0 * 64;
    #pragma unroll
    for (int i = 0; i < 4; ++i) {
      int idx = tid + i * 256;                     // 1024 uint4 = 128 rows x 8
      int row = idx >> 3, q = (idx & 7) << 3;
      *(uint4*)(Tdst + row * 64 + q) = *(const uint4*)(sOutT + row * 72 + q);
    }
    __syncthreads();
  }
}

// Stage 2: Y[m2,(m1,c)] = sum_w Ar[m2,w]*Tr'[w,(m1,c)] + Ai[m2,w]*Tin'[w,(m1,c)]
// (= Tr Ar^T - Ti Ai^T).  gridDim.x = 217: blocks with m1 >= KOUT are dead and
// never launched. Each block owns one m1 -> contiguous f32 output span.
__global__ __launch_bounds__(256) void spool_stage2(
    const unsigned short* __restrict__ Ar,
    const unsigned short* __restrict__ Ai,
    const unsigned short* __restrict__ Tr,
    const unsigned short* __restrict__ Tin,
    float* __restrict__ Y, int bBase, size_t zStride) {
  __shared__ unsigned short sAr[128 * LDST];
  __shared__ unsigned short sAi[128 * LDST];
  __shared__ unsigned short sTr[64  * LDST];
  __shared__ unsigned short sTi[64  * LDST];
  __shared__ float sOutY[128 * 64];                // output staging (32 KB)

  const size_t zoff = (size_t)blockIdx.z * zStride;
  Tr += zoff; Tin += zoff;
  const int b = bBase + blockIdx.z;

  const int tid = threadIdx.x;
  const int m1  = blockIdx.x;          // < KOUT by grid construction
  const int n0  = m1 * 64;
  const int m0  = blockIdx.y * 128;    // m2 tile (padded 256)
  const int wm  = (tid >> 5) >> 1;
  const int wn  = (tid >> 5) & 1;

  v8f acc[2][2] = {};

  for (int k0 = 0; k0 < 256; k0 += 32) {
    #pragma unroll
    for (int i = 0; i < 2; ++i) {                  // A tiles
      int idx = tid + i * 256;
      int row = idx >> 2, col = (idx & 3) << 3;
      *(uint4*)(sAr + row * LDST + col) = *(const uint4*)(Ar + (m0 + row) * 256 + k0 + col);
      *(uint4*)(sAi + row * LDST + col) = *(const uint4*)(Ai + (m0 + row) * 256 + k0 + col);
    }
    #pragma unroll
    for (int i = 0; i < 2; ++i) {                  // B tiles 32(k) x 64(n) bf16, transpose
      int idx = tid + i * 256;                     // 0..255 -> Tr, 256..511 -> Tin
      int mat = idx >> 8, sub = idx & 255;
      int kk = sub >> 3, nn = (sub & 7) << 3;
      const unsigned short* src = (mat ? Tin : Tr) + (size_t)(k0 + kk) * WC + n0 + nn;
      union { uint4 q; unsigned short h[8]; } u; u.q = *(const uint4*)src;
      unsigned short* dst = mat ? sTi : sTr;
      #pragma unroll
      for (int j = 0; j < 8; ++j) dst[(nn + j) * LDST + kk] = u.h[j];
    }
    __syncthreads();

    v16bf arf[2], aif[2], trf[2], tif[2];
    arf[0] = load_frag(sAr, wm * 32);      arf[1] = load_frag(sAr, wm * 32 + 16);
    aif[0] = load_frag(sAi, wm * 32);      aif[1] = load_frag(sAi, wm * 32 + 16);
    trf[0] = load_frag(sTr, wn * 32);      trf[1] = load_frag(sTr, wn * 32 + 16);
    tif[0] = load_frag(sTi, wn * 32);      tif[1] = load_frag(sTi, wn * 32 + 16);

    #pragma unroll
    for (int ms = 0; ms < 2; ++ms)
      #pragma unroll
      for (int ns = 0; ns < 2; ++ns) {
        acc[ms][ns] = __builtin_amdgcn_wmma_f32_16x16x32_bf16(
            false, arf[ms], false, trf[ns], (short)0, acc[ms][ns], false, false);
        acc[ms][ns] = __builtin_amdgcn_wmma_f32_16x16x32_bf16(
            false, aif[ms], false, tif[ns], (short)0, acc[ms][ns], false, false);
      }
    __syncthreads();
  }

  // Epilogue: acc -> LDS -> coalesced b128 stores of contiguous output rows
  // m2 in [m0, min(217, m0+128)).
  const int lane = tid & 31;
  const int rAdd = (lane >> 4) << 3;
  #pragma unroll
  for (int ms = 0; ms < 2; ++ms)
    #pragma unroll
    for (int ns = 0; ns < 2; ++ns) {
      int mloc = wm * 32 + ms * 16 + rAdd;
      int c    = wn * 32 + ns * 16 + (lane & 15);
      #pragma unroll
      for (int r = 0; r < 8; ++r)
        sOutY[(mloc + r) * 64 + c] = acc[ms][ns][r];
    }
  __syncthreads();

  const int nrows = (m0 + 128 <= KOUT) ? 128 : (KOUT - m0);
  float* dst = Y + (((size_t)b * KOUT + m1) * KOUT + m0) * 64;
  for (int idx = tid; idx < nrows * 16; idx += 256) {  // 16 uint4 per 64-f32 row
    int row = idx >> 4, q = (idx & 15) << 2;
    *(uint4*)(dst + row * 64 + q) = *(const uint4*)(sOutY + row * 64 + q);
  }
}

extern "C" void kernel_launch(void* const* d_in, const int* in_sizes, int n_in,
                              void* d_out, int out_size, void* d_ws, size_t ws_size,
                              hipStream_t stream) {
  const float* X = (const float*)d_in[0];          // (16,256,256,64) f32
  float* out = (float*)d_out;                      // (16,217,217,64) f32

  const size_t slice = (size_t)256 * WC;           // elements per batch (both X and T)
  unsigned short* Ar = (unsigned short*)d_ws;
  unsigned short* Ai = Ar + 256 * 256;
  unsigned short* T0 = Ai + 256 * 256;

  build_dft_mats<<<256, 256, 0, stream>>>(Ar, Ai);

  const size_t need_fused = (size_t)2 * 256 * 256 * 2 + (size_t)2 * 16 * slice * 2;
  if (ws_size >= need_fused) {
    // Batch-fused: T for all 16 batches resident in ws (~269 MB), 3 kernels total.
    unsigned short* Tr  = T0;
    unsigned short* Tin = T0 + 16 * slice;
    spool_stage1<<<dim3(256, 2, 16), 256, 0, stream>>>(X, Ar, Ai, Tr, Tin, slice);
    spool_stage2<<<dim3(KOUT, 2, 16), 256, 0, stream>>>(Ar, Ai, Tr, Tin, out, 0, slice);
  } else {
    // Per-batch fallback (~17 MB ws): stream order serializes stage1 -> stage2.
    unsigned short* Tr  = T0;
    unsigned short* Tin = T0 + slice;
    for (int b = 0; b < 16; ++b) {
      spool_stage1<<<dim3(256, 2, 1), 256, 0, stream>>>(X + (size_t)b * slice,
                                                        Ar, Ai, Tr, Tin, 0);
      spool_stage2<<<dim3(KOUT, 2, 1), 256, 0, stream>>>(Ar, Ai, Tr, Tin, out, b, 0);
    }
  }
}
